// MultiplicativeSelfAttention_2723009266485
// MI455X (gfx1250) — matrix operand order repaired and checked by reading, not compiled, not run on hardware
//
#include <hip/hip_runtime.h>

// MultiplicativeSelfAttention for MI455X (gfx1250, wave32, WMMA bf16).
// B=8, S=2048, D=768. bf16 WMMA with fp32 accum, flash-style fused softmax.
// Attention kernel: wave-pairs split D (no redundant matmul work) to cut
// register pressure (~280 VGPRs/wave) -> higher occupancy + load hoisting.

#define BATCH 8
#define SEQ   2048
#define DIM   768

typedef __attribute__((ext_vector_type(16))) __bf16 v16bf;
typedef __attribute__((ext_vector_type(8)))  float  v8f;

union FragU {
    v16bf v;
    uint4 q[2];
    unsigned short s[16];
};

union V8U {
    v8f  v;
    uint4 q[2];
};

// fp32 -> bf16 round-to-nearest-even
static __device__ inline unsigned short f2bf(float f) {
    unsigned int u = __float_as_uint(f);
    unsigned int r = u + 0x7FFFu + ((u >> 16) & 1u);
    return (unsigned short)(r >> 16);
}

static __device__ inline v8f wmma_bf16(v16bf a, v16bf b, v8f c) {
    // D = A(16x32 bf16) x B(32x16 bf16) + C(16x16 f32)
    return __builtin_amdgcn_wmma_f32_16x16x32_bf16(
        false, a, false, b, (short)0, c, false, false);
}

// B-fragment: 32 contiguous bytes (16 bf16, K-major) per lane
static __device__ inline v16bf ldfragB(const unsigned short* p) {
    FragU f;
    const uint4* q = (const uint4*)p;
    f.q[0] = q[0];
    f.q[1] = q[1];
    return f.v;
}

// A-fragment: two 16-byte chunks per lane
static __device__ inline v16bf ldfragA(const unsigned short* p) {
    FragU f;
    f.q[0] = *(const uint4*)p;
    f.q[1] = *(const uint4*)(p + 8);
    return f.v;
}

// ---------------------------------------------------------------------------
// Kernel 1: x -> xb (bf16 [B,S,D]) and xT (bf16 [B,D,S]) via tiled transpose.
// ---------------------------------------------------------------------------
__global__ __launch_bounds__(256) void prep_x_kernel(
    const float* __restrict__ x,
    unsigned short* __restrict__ xb,
    unsigned short* __restrict__ xT)
{
    __shared__ float tile[32][33];
    const int d0 = blockIdx.x * 32;
    const int s0 = blockIdx.y * 32;
    const int b  = blockIdx.z;
    const int tx = threadIdx.x, ty = threadIdx.y;
    const float* xB = x + (size_t)b * SEQ * DIM;
    unsigned short* xbB = xb + (size_t)b * SEQ * DIM;
    unsigned short* xTB = xT + (size_t)b * DIM * SEQ;

#pragma unroll
    for (int r = 0; r < 4; r++) {
        int s = s0 + ty + r * 8;
        float val = xB[(size_t)s * DIM + d0 + tx];
        tile[ty + r * 8][tx] = val;
        xbB[(size_t)s * DIM + d0 + tx] = f2bf(val);
    }
    __syncthreads();
#pragma unroll
    for (int r = 0; r < 4; r++) {
        int d = d0 + ty + r * 8;
        xTB[(size_t)d * SEQ + s0 + tx] = f2bf(tile[tx][ty + r * 8]);
    }
}

// ---------------------------------------------------------------------------
// Kernel 2: convert W, proj_w -> bf16
// ---------------------------------------------------------------------------
__global__ __launch_bounds__(256) void prep_w_kernel(
    const float* __restrict__ W, const float* __restrict__ pw,
    unsigned short* __restrict__ Wb, unsigned short* __restrict__ pwb)
{
    int i = blockIdx.x * 256 + threadIdx.x;
    if (i < DIM * DIM) {
        Wb[i]  = f2bf(W[i]);
        pwb[i] = f2bf(pw[i]);
    }
}

// ---------------------------------------------------------------------------
// Kernel 3: C[M,768] = A[M,768] @ Bm[768,768]^T  (bf16 in, bf16 out)
// Each wave: 16 rows x 64 cols (4 C tiles). 12288 waves, 4 waves/block.
// ---------------------------------------------------------------------------
__global__ __launch_bounds__(128) void gemm_bf16_kernel(
    const unsigned short* __restrict__ A,
    const unsigned short* __restrict__ Bm,
    unsigned short* __restrict__ Cout)
{
    const int wid    = blockIdx.x * 4 + (threadIdx.x >> 5);
    const int lane   = threadIdx.x & 31;
    const int lane16 = lane & 15;
    const int half   = lane >> 4;
    const int mT = wid / 12;
    const int nS = wid % 12;

    const unsigned short* aRow = A + (size_t)(mT * 16 + lane16) * DIM + half * 16;

    v8f c[4];
#pragma unroll
    for (int t = 0; t < 4; t++)
        c[t] = (v8f){0.f, 0.f, 0.f, 0.f, 0.f, 0.f, 0.f, 0.f};

#pragma unroll
    for (int k = 0; k < DIM / 32; k++) {
        v16bf a = ldfragA(aRow + k * 32);
#pragma unroll
        for (int t = 0; t < 4; t++) {
            const unsigned short* bRow =
                Bm + (size_t)(nS * 64 + t * 16 + lane16) * DIM + half * 16 + k * 32;
            c[t] = wmma_bf16(a, ldfragB(bRow), c[t]);
        }
    }
#pragma unroll
    for (int t = 0; t < 4; t++) {
#pragma unroll
        for (int v = 0; v < 8; v++) {
            int r   = mT * 16 + v + half * 8;
            int col = nS * 64 + t * 16 + lane16;
            Cout[(size_t)r * DIM + col] = f2bf(c[t][v]);
        }
    }
}

// ---------------------------------------------------------------------------
// Kernel 4: fused flash attention core (D-split wave pairs).
//   s[i,j] = sum_e Wx[i,e]*x[j,e]; mask diag; online softmax; c += p @ x_j.
// Block: 128 threads = 4 waves = 2 pairs; each pair owns one 16-row i-tile.
// Within a pair, wave dh owns d-range [dh*384, dh*384+384):
//   - partial s over its d-half (24 WMMA), summed with partner via LDS
//   - c-update for its 24 d-tiles (24 WMMA), 24 v8f accumulators.
// grid (S/32, B).
// ---------------------------------------------------------------------------
__global__ __launch_bounds__(128) void attn_kernel(
    const unsigned short* __restrict__ xb,   // [B,S,D]
    const unsigned short* __restrict__ xT,   // [B,D,S]
    const unsigned short* __restrict__ Wxb,  // [B,S,D]
    unsigned short* __restrict__ cb)         // [B,S,D]
{
    __shared__ __align__(16) unsigned short ldsWx[32 * DIM];  // 48 KB
    __shared__ __align__(16) float ldsX[4][512];              // 8 KB: s-exchange, reused as p scratch

    const int b    = blockIdx.y;
    const int i0   = blockIdx.x * 32;
    const int tid  = threadIdx.x;
    const int wave = tid >> 5;
    const int lane = tid & 31;
    const int lane16 = lane & 15;
    const int half   = lane >> 4;
    const int pairI  = wave >> 1;   // i-tile within block (0,1)
    const int dh     = wave & 1;    // d-half (0,1)

    // cooperative load of the Wx i-block into LDS (32 x 768 bf16)
    {
        const uint4* src = (const uint4*)(Wxb + (size_t)b * SEQ * DIM + (size_t)i0 * DIM);
        uint4* dst = (uint4*)ldsWx;
        for (int it = tid; it < (32 * DIM * 2) / 16; it += 128)
            dst[it] = src[it];
    }
    __syncthreads();

    const unsigned short* xbB = xb + (size_t)b * SEQ * DIM;
    const unsigned short* xTB = xT + (size_t)b * DIM * SEQ;
    const int dbase = dh * (DIM / 2);        // 0 or 384
    const int tbase = dh * 24;               // first d-tile index owned

    v8f acc[24];
#pragma unroll
    for (int t = 0; t < 24; t++)
        acc[t] = (v8f){0.f, 0.f, 0.f, 0.f, 0.f, 0.f, 0.f, 0.f};

    float mrow[8], lrow[8];
#pragma unroll
    for (int v = 0; v < 8; v++) { mrow[v] = -3.0e38f; lrow[v] = 0.f; }

    // A-fragment base for s-GEMM: Wx rows from LDS, this wave's d-half
    const unsigned short* aRow =
        ldsWx + (size_t)(pairI * 16 + lane16) * DIM + dbase + half * 16;
    float* myS   = &ldsX[wave][0];
    const float* prS = &ldsX[wave ^ 1][0];
    const int ig_base = i0 + pairI * 16 + half * 8;

#pragma unroll 1
    for (int j0 = 0; j0 < SEQ; j0 += 32) {
        // --- partial s tile (16 x 32) over this wave's 384-wide d-half ---
        v8f s0 = (v8f){0.f, 0.f, 0.f, 0.f, 0.f, 0.f, 0.f, 0.f};
        v8f s1 = (v8f){0.f, 0.f, 0.f, 0.f, 0.f, 0.f, 0.f, 0.f};
        const unsigned short* b0Row = xbB + (size_t)(j0 + lane16) * DIM + dbase + half * 16;
        const unsigned short* b1Row = xbB + (size_t)(j0 + 16 + lane16) * DIM + dbase + half * 16;
#pragma unroll
        for (int k = 0; k < 12; k++) {
            v16bf a = ldfragA(aRow + k * 32);
            s0 = wmma_bf16(a, ldfragB(b0Row + k * 32), s0);
            s1 = wmma_bf16(a, ldfragB(b1Row + k * 32), s1);
        }

        // --- exchange partials with partner wave through LDS, sum ---
        {
            V8U w0; w0.v = s0;
            V8U w1; w1.v = s1;
            uint4* pw = (uint4*)(myS + lane * 16);
            pw[0] = w0.q[0]; pw[1] = w0.q[1];
            pw[2] = w1.q[0]; pw[3] = w1.q[1];
        }
        __syncthreads();
        {
            const uint4* pr = (const uint4*)(prS + lane * 16);
            V8U o0, o1;
            o0.q[0] = pr[0]; o0.q[1] = pr[1];
            o1.q[0] = pr[2]; o1.q[1] = pr[3];
            s0 = s0 + o0.v;
            s1 = s1 + o1.v;
        }
        __syncthreads();   // partner done reading before we reuse myS for p

        // --- diagonal mask (i == j -> -inf) ---
#pragma unroll
        for (int v = 0; v < 8; v++) {
            int ig = ig_base + v;
            s0[v] = (ig == j0 + lane16)      ? -3.0e38f : s0[v];
            s1[v] = (ig == j0 + 16 + lane16) ? -3.0e38f : s1[v];
        }

        // --- online softmax (redundant in both waves of the pair) ---
        float scale[8];
#pragma unroll
        for (int v = 0; v < 8; v++) {
            float m = fmaxf(s0[v], s1[v]);
            m = fmaxf(m, __shfl_xor(m, 1));
            m = fmaxf(m, __shfl_xor(m, 2));
            m = fmaxf(m, __shfl_xor(m, 4));
            m = fmaxf(m, __shfl_xor(m, 8));
            float mn = fmaxf(mrow[v], m);
            scale[v] = __expf(mrow[v] - mn);
            mrow[v]  = mn;
            float p0 = __expf(s0[v] - mn);
            float p1 = __expf(s1[v] - mn);
            s0[v] = p0;
            s1[v] = p1;
            float rs = p0 + p1;
            rs += __shfl_xor(rs, 1);
            rs += __shfl_xor(rs, 2);
            rs += __shfl_xor(rs, 4);
            rs += __shfl_xor(rs, 8);
            lrow[v] = lrow[v] * scale[v] + rs;
        }

        // --- transpose p (C-layout) -> A-layout via own LDS scratch ---
#pragma unroll
        for (int v = 0; v < 8; v++) {
            int r = v + half * 8;
            myS[r * 32 + lane16]      = s0[v];
            myS[r * 32 + 16 + lane16] = s1[v];
        }
        // same-wave LDS ops stay in order: no barrier needed
        v16bf pa;
        {
            const float* pr = myS + lane16 * 32 + half * 16;
            FragU fb;
#pragma unroll
            for (int i = 0; i < 8; i++) {
                fb.s[i]     = f2bf(pr[i]);        // K 0..7  / 16..23
                fb.s[8 + i] = f2bf(pr[8 + i]);    // K 8..15 / 24..31
            }
            pa = fb.v;
        }

        // --- c-update for this wave's 24 d-tiles; rescale fused per tile ---
        const unsigned short* xTj =
            xTB + (size_t)(tbase * 16 + lane16) * SEQ + j0 + half * 16;
#pragma unroll
        for (int t = 0; t < 24; t++) {
            v16bf bb = ldfragB(xTj + (size_t)t * 16 * SEQ);
#pragma unroll
            for (int v = 0; v < 8; v++)
                acc[t][v] *= scale[v];
            acc[t] = wmma_bf16(pa, bb, acc[t]);
        }
    }

    // --- epilogue: normalize and store bf16 ---
    float linv[8];
#pragma unroll
    for (int v = 0; v < 8; v++) linv[v] = 1.0f / lrow[v];

    unsigned short* cB = cb + (size_t)b * SEQ * DIM;
#pragma unroll
    for (int t = 0; t < 24; t++) {
#pragma unroll
        for (int v = 0; v < 8; v++) {
            int r   = i0 + pairI * 16 + v + half * 8;
            int col = (tbase + t) * 16 + lane16;
            cB[(size_t)r * DIM + col] = f2bf(acc[t][v] * linv[v]);
        }
    }
}

// ---------------------------------------------------------------------------
// Kernel 5: out = x + relu(cb @ proj_w^T + bias)   (fp32 out)
// ---------------------------------------------------------------------------
__global__ __launch_bounds__(128) void gemm_proj_kernel(
    const unsigned short* __restrict__ A,    // cb bf16 [B*S, D]
    const unsigned short* __restrict__ Bm,   // proj_w bf16 [D, D]
    const float* __restrict__ bias,
    const float* __restrict__ x,
    float* __restrict__ out)
{
    const int wid    = blockIdx.x * 4 + (threadIdx.x >> 5);
    const int lane   = threadIdx.x & 31;
    const int lane16 = lane & 15;
    const int half   = lane >> 4;
    const int mT = wid / 12;
    const int nS = wid % 12;

    const unsigned short* aRow = A + (size_t)(mT * 16 + lane16) * DIM + half * 16;

    v8f c[4];
#pragma unroll
    for (int t = 0; t < 4; t++)
        c[t] = (v8f){0.f, 0.f, 0.f, 0.f, 0.f, 0.f, 0.f, 0.f};

#pragma unroll
    for (int k = 0; k < DIM / 32; k++) {
        v16bf a = ldfragA(aRow + k * 32);
#pragma unroll
        for (int t = 0; t < 4; t++) {
            const unsigned short* bRow =
                Bm + (size_t)(nS * 64 + t * 16 + lane16) * DIM + half * 16 + k * 32;
            c[t] = wmma_bf16(a, ldfragB(bRow), c[t]);
        }
    }
#pragma unroll
    for (int t = 0; t < 4; t++) {
        float bval = bias[nS * 64 + t * 16 + lane16];
#pragma unroll
        for (int v = 0; v < 8; v++) {
            int r   = mT * 16 + v + half * 8;
            int col = nS * 64 + t * 16 + lane16;
            float val = fmaxf(c[t][v] + bval, 0.f);
            out[(size_t)r * DIM + col] = x[(size_t)r * DIM + col] + val;
        }
    }
}

// ---------------------------------------------------------------------------
extern "C" void kernel_launch(void* const* d_in, const int* in_sizes, int n_in,
                              void* d_out, int out_size, void* d_ws, size_t ws_size,
                              hipStream_t stream) {
    const float* x  = (const float*)d_in[0];   // [8,2048,768]
    const float* W  = (const float*)d_in[1];   // [768,768]
    const float* pw = (const float*)d_in[2];   // [768,768]
    const float* pb = (const float*)d_in[3];   // [768]
    float* out = (float*)d_out;

    // workspace layout (bf16 buffers), total ~98.2 MiB
    char* ws = (char*)d_ws;
    const size_t SB = (size_t)BATCH * SEQ * DIM * 2;  // 25,165,824 B
    unsigned short* xb  = (unsigned short*)(ws);
    unsigned short* xT  = (unsigned short*)(ws + SB);
    unsigned short* Wxb = (unsigned short*)(ws + 2 * SB);
    unsigned short* cb  = (unsigned short*)(ws + 3 * SB);
    unsigned short* Wb  = (unsigned short*)(ws + 4 * SB);
    unsigned short* pwb = (unsigned short*)(ws + 4 * SB + (size_t)DIM * DIM * 2);

    prep_x_kernel<<<dim3(DIM / 32, SEQ / 32, BATCH), dim3(32, 8), 0, stream>>>(x, xb, xT);
    prep_w_kernel<<<(DIM * DIM + 255) / 256, 256, 0, stream>>>(W, pw, Wb, pwb);
    gemm_bf16_kernel<<<(BATCH * SEQ / 16) * (DIM / 64) / 4, 128, 0, stream>>>(xb, Wb, Wxb);
    attn_kernel<<<dim3(SEQ / 32, BATCH), 128, 0, stream>>>(xb, xT, Wxb, cb);
    gemm_proj_kernel<<<(BATCH * SEQ / 16) * (DIM / 64) / 4, 128, 0, stream>>>(cb, pwb, pb, x, out);
}
